// IRFDeconv3D_43963285242216
// MI455X (gfx1250) — matrix operand order repaired and checked by reading, not compiled, not run on hardware
//
#include <hip/hip_runtime.h>

// CDNA5 (gfx1250) wave32 WMMA types
typedef __attribute__((ext_vector_type(16))) _Float16 v16h;
typedef __attribute__((ext_vector_type(8)))  float    v8f;
typedef __attribute__((ext_vector_type(4)))  float    v4f;

#define T_BINS        256
#define PIX_PER_BATCH (128 * 128)
#define PIX_PER_BLOCK 128          // 8 waves * 16 pixels
#define NUM_BATCH     8

// One M-tile step of the triangular Toeplitz GEMM.
// acc(16x16) += sum_{kt: 2kt<=MT} A[dlt=MT-2kt] (16x32) * B[kt] (32x16)
// A fragments live in LDS in exact WMMA A-operand layout; B fragments are in
// registers (constant indices after full unroll).
template <int MT>
__device__ __forceinline__ void mt_step(const _Float16* s_afrag, int lane,
                                        const v16h (&bf)[8], float* obase) {
    v8f acc = {};
#pragma unroll
    for (int kt = 0; kt <= (MT >> 1); ++kt) {
        const int dlt = MT - 2 * kt;
        v16h av = *(const v16h*)(s_afrag + (dlt * 32 + lane) * 16);
        acc = __builtin_amdgcn_wmma_f32_16x16x32_f16(
            /*neg_a=*/false, av, /*neg_b=*/false, bf[kt],
            /*c_mod=*/(short)0, acc, /*reuse_a=*/false, /*reuse_b=*/false);
    }
    // D layout: lane holds column N=lane&15; VGPR r holds row M = r + (hi?8:0).
    const int hi = lane >> 4;
    float* o = obase + MT * 16 + hi * 8;   // 8 consecutive time bins
    v4f d0 = {acc[0], acc[1], acc[2], acc[3]};
    v4f d1 = {acc[4], acc[5], acc[6], acc[7]};
    // Output is written once and never re-read: non-temporal stores keep the
    // streaming writes from occupying L2 (pure HBM-bandwidth-bound kernel).
    __builtin_nontemporal_store(d0, (v4f*)(o));
    __builtin_nontemporal_store(d1, (v4f*)(o + 4));
}

__global__ __launch_bounds__(256)
void irf_deconv_wmma(const float* __restrict__ data,
                     const float* __restrict__ irf,
                     float* __restrict__ out) {
    __shared__ float               s_irf[T_BINS];
    __shared__ float               s_red[T_BINS];
    __shared__ _Float16            s_kn[T_BINS];
    __shared__ __attribute__((aligned(32))) _Float16 s_afrag[16 * 32 * 16]; // 16 KB

    const int tid  = threadIdx.x;
    const int bb   = blockIdx.x >> 7;    // batch index (8)
    const int tile = blockIdx.x & 127;   // 128-pixel tile within batch

    // ---- stage 1: per-batch kernel normalization kn = irf / max(irf) ----
    {
        float v = irf[bb * T_BINS + tid];
        s_irf[tid] = v;
        s_red[tid] = v;
        __syncthreads();
#pragma unroll
        for (int s = 128; s > 0; s >>= 1) {
            if (tid < s) s_red[tid] = fmaxf(s_red[tid], s_red[tid + s]);
            __syncthreads();
        }
        const float inv_max = 1.0f / s_red[0];
        s_kn[tid] = (_Float16)(s_irf[tid] * inv_max);
        __syncthreads();
    }

    // ---- stage 2: build the 16 distinct Toeplitz A-fragments in LDS ----
    // A-tile(mt,kt) element (i,j) = kn[16*dlt + i - j], dlt = mt - 2kt in [0,15].
    // Stored per-lane in the ISA's 16-bit A (16x32) VGPR layout:
    //   lane = M (0..15) | lanes 16-31 same M, upper-K half
    //   element e=2v+h -> K = 2*(v&3)+h + (v>=4?16:0) + (hi?8:0)
#pragma unroll
    for (int q = 0; q < 2; ++q) {
        const int slot = tid * 2 + q;        // 512 lane-slots total
        const int dlt  = slot >> 5;
        const int lane = slot & 31;
        const int hi   = lane >> 4;
        const int M    = lane & 15;
#pragma unroll
        for (int e = 0; e < 16; ++e) {
            const int vreg = e >> 1, h = e & 1;
            const int K = 2 * (vreg & 3) + h + ((vreg >= 4) ? 16 : 0) + (hi ? 8 : 0);
            const int d = 16 * dlt + M - K;
            s_afrag[slot * 16 + e] =
                (d >= 0 && d < T_BINS) ? s_kn[d] : (_Float16)0.0f;
        }
    }
    __syncthreads();

    // ---- stage 3: triangular Toeplitz GEMM, 16 pixels per wave ----
    const int wave = tid >> 5;
    const int lane = tid & 31;
    const int hi   = lane >> 4;
    const int pix  = tile * PIX_PER_BLOCK + wave * 16 + (lane & 15);

    const float* dbase = data + ((size_t)(bb * PIX_PER_BATCH + pix)) * T_BINS;
    float*       obase = out  + ((size_t)(bb * PIX_PER_BATCH + pix)) * T_BINS;

    // Load all 8 B-fragments (full K=256) for this wave's 16 pixels.
    // B layout: element s holds K = (s<8 ? s : s+8) + (hi?8:0), column N = lane&15.
    // => each lane reads 8 bins at 32*kt + hi*8 and 8 bins at 32*kt + 16 + hi*8.
    // Each input float is consumed exactly once -> non-temporal loads.
    v16h bf[8];
#pragma unroll
    for (int kt = 0; kt < 8; ++kt) {
        const int t0 = 32 * kt + hi * 8;
        const v4f* s4 = (const v4f*)(dbase + t0);
        v16h bv;
#pragma unroll
        for (int i = 0; i < 4; ++i) {
            v4f f = __builtin_nontemporal_load(s4 + i + (i >> 1) * 2);
            bv[4 * i + 0] = (_Float16)f.x;
            bv[4 * i + 1] = (_Float16)f.y;
            bv[4 * i + 2] = (_Float16)f.z;
            bv[4 * i + 3] = (_Float16)f.w;
        }
        bf[kt] = bv;
    }

    // 72 WMMAs per wave (zero tiles of the lower-triangular operand skipped).
#define MT_STEP(MT) mt_step<MT>(s_afrag, lane, bf, obase)
    MT_STEP(0);  MT_STEP(1);  MT_STEP(2);  MT_STEP(3);
    MT_STEP(4);  MT_STEP(5);  MT_STEP(6);  MT_STEP(7);
    MT_STEP(8);  MT_STEP(9);  MT_STEP(10); MT_STEP(11);
    MT_STEP(12); MT_STEP(13); MT_STEP(14); MT_STEP(15);
#undef MT_STEP
}

extern "C" void kernel_launch(void* const* d_in, const int* in_sizes, int n_in,
                              void* d_out, int out_size, void* d_ws, size_t ws_size,
                              hipStream_t stream) {
    (void)in_sizes; (void)n_in; (void)out_size; (void)d_ws; (void)ws_size;
    const float* data = (const float*)d_in[0];  // [8,128,128,256,1] f32
    const float* irf  = (const float*)d_in[1];  // [8,1,1,256,1]     f32
    float*       out  = (float*)d_out;          // [8,128,128,256,1] f32

    const dim3 grid(NUM_BATCH * (PIX_PER_BATCH / PIX_PER_BLOCK)); // 1024 blocks
    const dim3 block(256);                                        // 8 waves
    irf_deconv_wmma<<<grid, block, 0, stream>>>(data, irf, out);
}
